// MultiheadAttention_1159641170002
// MI455X (gfx1250) — matrix operand order repaired and checked by reading, not compiled, not run on hardware
//
#include <hip/hip_runtime.h>

// MultiheadAttention forward for MI455X (gfx1250), wave32 + WMMA f16.
// d_out = [ out: T*B*E f32 | avg_weights: B*T*T f32 ]
// Workspace layout (all f16 except noted):
//   Qh [B,H,T,64] | Kh [B,H,T,64] | Vt [B,H,64,T] | stag f16 [T,B,E]
//   Xq/Xk/Xv f16 [M,E] | Wh f16 [3E,E] | Woh f16 [E,E]          (~93 MB)

#define T_DIM 2048
#define B_DIM 4
#define E_DIM 768
#define H_DIM 12
#define HD_DIM 64
#define M_DIM (T_DIM * B_DIM)      // 8192
#define SCALING 0.125f             // HD^-0.5
#define INV_H (1.0f / 12.0f)

typedef __attribute__((ext_vector_type(16))) _Float16 v16h;
typedef __attribute__((ext_vector_type(8)))  _Float16 v8h;
typedef __attribute__((ext_vector_type(4)))  _Float16 v4h;
typedef __attribute__((ext_vector_type(8)))  float    v8f;

__device__ __forceinline__ v8f wmma_f16(v16h a, v16h b, v8f c) {
  // D = A(16x32 f16) x B(32x16 f16) + C(16x16 f32)
  return __builtin_amdgcn_wmma_f32_16x16x32_f16(false, a, false, b, (short)0, c,
                                                false, false);
}

// ---- fragment loaders (wave32 layouts per CDNA5 ISA 7.12.2) -------------
// A 16x32 f16: lane L -> M = L&15, g = L>>4; halves 0..7 -> K = g*8+i,
// halves 8..15 -> K = 16+g*8+(i-8)  => two contiguous 16B chunks.
__device__ __forceinline__ v16h load_a_f16(const _Float16* __restrict__ rowp, int g) {
  const v8h lo = *(const v8h*)(rowp + g * 8);
  const v8h hi = *(const v8h*)(rowp + 16 + g * 8);
  v16h a;
#pragma unroll
  for (int i = 0; i < 8; ++i) { a[i] = lo[i]; a[8 + i] = hi[i]; }
  return a;
}
// B 32x16 f16: lane L -> N = L&15, g = L>>4; halves 0..7 -> K = g*8+i,
// halves 8..15 -> K = 16+g*8+(i-8)  => two contiguous 16B chunks.
__device__ __forceinline__ v16h load_b_f16(const _Float16* __restrict__ colp, int g) {
  const v8h lo = *(const v8h*)(colp + g * 8);
  const v8h hi = *(const v8h*)(colp + 16 + g * 8);
  v16h b;
#pragma unroll
  for (int i = 0; i < 8; ++i) { b[i] = lo[i]; b[8 + i] = hi[i]; }
  return b;
}

// reductions across the 16 lanes holding one C/D row (offsets 1,2,4,8 stay in-group)
__device__ __forceinline__ float rmax16(float x) {
#pragma unroll
  for (int o = 1; o < 16; o <<= 1) x = fmaxf(x, __shfl_xor(x, o, 32));
  return x;
}
__device__ __forceinline__ float rsum16(float x) {
#pragma unroll
  for (int o = 1; o < 16; o <<= 1) x += __shfl_xor(x, o, 32);
  return x;
}

// ================= Kernel 0: bulk f32 -> f16 conversion ====================
__global__ __launch_bounds__(256) void cvt_f16_kernel(
    const float* __restrict__ src, _Float16* __restrict__ dst, int n4) {
  int i = blockIdx.x * 256 + threadIdx.x;
  const int stride = gridDim.x * 256;
  for (; i < n4; i += stride) {
    const float4 f = ((const float4*)src)[i];
    v4h h;
    h[0] = (_Float16)f.x; h[1] = (_Float16)f.y;
    h[2] = (_Float16)f.z; h[3] = (_Float16)f.w;
    ((v4h*)dst)[i] = h;
  }
}

// ======================= Kernel 1: fused QKV projection ====================
// Wave computes a 32x64 tile: out[m,e] = dot(Xh[m,:], Wh[e,:]) + bias[e].
// Q pre-scaled; layouts: Qh/Kh [B,H,T,64] f16, Vt [B,H,64,T] f16.
__global__ __launch_bounds__(128) void qkv_proj_kernel(
    const _Float16* __restrict__ Xq, const _Float16* __restrict__ Xk,
    const _Float16* __restrict__ Xv, const _Float16* __restrict__ Wh,
    const float* __restrict__ bias, _Float16* __restrict__ Qh,
    _Float16* __restrict__ Kh, _Float16* __restrict__ Vt) {
  const int lane = threadIdx.x & 31;
  const int wave = threadIdx.x >> 5;
  const int g = lane >> 4;
  const int n = lane & 15;
  const int row0 = blockIdx.x * 32;
  const int colblk = blockIdx.y * 4 + wave;   // 0..35 (64 cols each of 3E)
  const int e0 = colblk * 64;
  const int seg = e0 / E_DIM;                 // 0:q 1:k 2:v
  const int es0 = e0 - seg * E_DIM;
  const _Float16* X = (seg == 0) ? Xq : (seg == 1) ? Xk : Xv;
  const _Float16* arow0 = X + (size_t)(row0 + n) * E_DIM;
  const _Float16* arow1 = arow0 + (size_t)16 * E_DIM;

  v8f acc0[4] = {}, acc1[4] = {};
  for (int kb = 0; kb < E_DIM; kb += 32) {
    const v16h a0 = load_a_f16(arow0 + kb, g);
    const v16h a1 = load_a_f16(arow1 + kb, g);
#pragma unroll
    for (int j = 0; j < 4; ++j) {
      const v16h b = load_b_f16(Wh + (size_t)(e0 + j * 16 + n) * E_DIM + kb, g);
      acc0[j] = wmma_f16(a0, b, acc0[j]);
      acc1[j] = wmma_f16(a1, b, acc1[j]);
    }
  }
#pragma unroll
  for (int tf = 0; tf < 2; ++tf) {
    const v8f* acc = tf ? acc1 : acc0;
#pragma unroll
    for (int j = 0; j < 4; ++j) {
      const int e = e0 + j * 16 + n;
      const int es = es0 + j * 16 + n;
      const int h = es >> 6, d = es & 63;
      const float bi = bias[e];
#pragma unroll
      for (int v = 0; v < 8; ++v) {
        const int m = row0 + tf * 16 + v + 8 * g;   // m = t*B + b
        const int t = m >> 2, bb = m & 3;
        const float val = acc[j][v] + bi;
        if (seg == 0)
          Qh[((size_t)(bb * H_DIM + h) * T_DIM + t) * HD_DIM + d] =
              (_Float16)(val * SCALING);
        else if (seg == 1)
          Kh[((size_t)(bb * H_DIM + h) * T_DIM + t) * HD_DIM + d] = (_Float16)val;
        else
          Vt[((size_t)(bb * H_DIM + h) * HD_DIM + d) * T_DIM + t] = (_Float16)val;
      }
    }
  }
}

// ======================= Kernel 2: attention ==============================
// grid (T/16, B); 128 threads = 4 waves; wave w handles heads w, w+4, w+8.
// Shared: avg tile [16][2048] f32 (ds_add_f32 cross-head) + per-wave W-transpose
// scratch (2 tiles of 16x17 f32).
__global__ __launch_bounds__(128) void attention_kernel(
    const _Float16* __restrict__ Qh, const _Float16* __restrict__ Kh,
    const _Float16* __restrict__ Vt, _Float16* __restrict__ stag,
    float* __restrict__ avg_out) {
  extern __shared__ float smem[];
  float* s_avg = smem;                         // 16*2048 floats (128 KB)
  float* s_scr = smem + 16 * 2048;             // 4 waves * 2 * 16*17 floats
  const int tid = threadIdx.x;
  const int lane = tid & 31, wave = tid >> 5;
  const int g = lane >> 4, n = lane & 15;
  const int q0 = blockIdx.x * 16;
  const int b = blockIdx.y;

  for (int i = tid; i < 16 * 2048; i += 128) s_avg[i] = 0.f;
  __syncthreads();

  float* scr0 = s_scr + wave * (2 * 16 * 17);
  float* scr1 = scr0 + 16 * 17;

  for (int r = 0; r < 3; ++r) {
    const int h = wave + 4 * r;
    const size_t hb = (size_t)(b * H_DIM + h);
    const _Float16* qrow = Qh + (hb * T_DIM + q0 + n) * HD_DIM;
    const v16h aq0 = load_a_f16(qrow, g);        // d 0..31
    const v16h aq1 = load_a_f16(qrow + 32, g);   // d 32..63
    const _Float16* kbase = Kh + hb * T_DIM * HD_DIM;
    const _Float16* vbase = Vt + hb * HD_DIM * T_DIM;

    // ---- pass 1: online row max / exp-sum over all S ----
    float mrun[8], lrun[8];
#pragma unroll
    for (int v = 0; v < 8; ++v) { mrun[v] = -3.0e38f; lrun[v] = 0.f; }

    for (int s0 = 0; s0 < T_DIM; s0 += 16) {
      const _Float16* krow = kbase + (size_t)(s0 + n) * HD_DIM;
      v8f c{};
      c = wmma_f16(aq0, load_b_f16(krow, g), c);
      c = wmma_f16(aq1, load_b_f16(krow + 32, g), c);
#pragma unroll
      for (int v = 0; v < 8; ++v) {
        const float tmax = rmax16(c[v]);
        const float mn = fmaxf(mrun[v], tmax);
        const float ts = rsum16(__expf(c[v] - mn));
        lrun[v] = lrun[v] * __expf(mrun[v] - mn) + ts;
        mrun[v] = mn;
      }
      if (s0 + 16 < T_DIM) __builtin_prefetch(krow + 16 * HD_DIM, 0, 0);
    }
    float linv[8];
#pragma unroll
    for (int v = 0; v < 8; ++v) linv[v] = 1.f / lrun[v];

    // ---- pass 2: weights -> avg (LDS atomics) + W x V (WMMA) ----
    v8f o[4] = {};
    for (int s0 = 0; s0 < T_DIM; s0 += 32) {
#pragma unroll
      for (int st = 0; st < 2; ++st) {
        const _Float16* krow = kbase + (size_t)(s0 + st * 16 + n) * HD_DIM;
        v8f c{};
        c = wmma_f16(aq0, load_b_f16(krow, g), c);
        c = wmma_f16(aq1, load_b_f16(krow + 32, g), c);
        float* scr = st ? scr1 : scr0;
#pragma unroll
        for (int v = 0; v < 8; ++v) {
          const int row = v + 8 * g;
          const float wgt = __expf(c[v] - mrun[v]) * linv[v];
          atomicAdd(&s_avg[row * 2048 + s0 + st * 16 + n], wgt * INV_H);
          scr[row * 17 + n] = wgt;               // stash for transpose
        }
      }
      asm volatile("s_wait_dscnt 0x0" ::: "memory");
      // rebuild W as an A fragment (lane = q-row n, K = s offset 0..31)
      v16h aw;
#pragma unroll
      for (int i = 0; i < 16; ++i) {
        const int K = (i < 8) ? (g * 8 + i) : (16 + g * 8 + (i - 8));
        const float val = (K < 16) ? scr0[n * 17 + K] : scr1[n * 17 + (K - 16)];
        aw[i] = (_Float16)val;
      }
#pragma unroll
      for (int j = 0; j < 4; ++j) {
        const _Float16* vrow = vbase + (size_t)(j * 16 + n) * T_DIM + s0;
        o[j] = wmma_f16(aw, load_b_f16(vrow, g), o[j]);
      }
    }

    // write this head's attn rows to f16 staging [T,B,E]
#pragma unroll
    for (int j = 0; j < 4; ++j) {
      const int e = h * HD_DIM + j * 16 + n;
#pragma unroll
      for (int v = 0; v < 8; ++v) {
        const int t = q0 + v + 8 * g;
        stag[((size_t)t * B_DIM + b) * E_DIM + e] = (_Float16)o[j][v];
      }
    }
  }
  __syncthreads();

  // flush the head-averaged weights tile: avg[b][q0+r][s]
  float* dst = avg_out + ((size_t)b * T_DIM + q0) * T_DIM;
  for (int i = tid * 4; i < 16 * 2048; i += 128 * 4) {
    const int rr = i >> 11;       // row within tile
    const int s = i & 2047;
    *(float4*)(dst + (size_t)rr * T_DIM + s) = *(const float4*)(s_avg + i);
  }
}

// ======================= Kernel 3: output projection =======================
// Wave computes a 32x64 tile from f16 staging x f16 out_w, f32 out + bias.
__global__ __launch_bounds__(128) void out_proj_kernel(
    const _Float16* __restrict__ attn, const _Float16* __restrict__ w,
    const float* __restrict__ bias, float* __restrict__ out) {
  const int lane = threadIdx.x & 31;
  const int wave = threadIdx.x >> 5;
  const int g = lane >> 4;
  const int n = lane & 15;
  const int row0 = blockIdx.x * 32;
  const int e0 = (blockIdx.y * 4 + wave) * 64;   // 0..704
  const _Float16* arow0 = attn + (size_t)(row0 + n) * E_DIM;
  const _Float16* arow1 = arow0 + (size_t)16 * E_DIM;

  v8f acc0[4] = {}, acc1[4] = {};
  for (int kb = 0; kb < E_DIM; kb += 32) {
    const v16h a0 = load_a_f16(arow0 + kb, g);
    const v16h a1 = load_a_f16(arow1 + kb, g);
#pragma unroll
    for (int j = 0; j < 4; ++j) {
      const v16h b = load_b_f16(w + (size_t)(e0 + j * 16 + n) * E_DIM + kb, g);
      acc0[j] = wmma_f16(a0, b, acc0[j]);
      acc1[j] = wmma_f16(a1, b, acc1[j]);
    }
  }
#pragma unroll
  for (int tf = 0; tf < 2; ++tf) {
    const v8f* acc = tf ? acc1 : acc0;
#pragma unroll
    for (int j = 0; j < 4; ++j) {
      const int e = e0 + j * 16 + n;
      const float bi = bias[e];
#pragma unroll
      for (int v = 0; v < 8; ++v) {
        const int m = row0 + tf * 16 + v + 8 * g;
        out[(size_t)m * E_DIM + e] = acc[j][v] + bi;
      }
    }
  }
}

// =========================================================================
extern "C" void kernel_launch(void* const* d_in, const int* in_sizes, int n_in,
                              void* d_out, int out_size, void* d_ws, size_t ws_size,
                              hipStream_t stream) {
  (void)in_sizes; (void)n_in; (void)out_size; (void)ws_size;
  const float* query = (const float*)d_in[0];
  const float* key   = (const float*)d_in[1];
  const float* value = (const float*)d_in[2];
  const float* w_in  = (const float*)d_in[3];
  const float* b_in  = (const float*)d_in[4];
  const float* w_out = (const float*)d_in[5];
  const float* b_out = (const float*)d_in[6];

  float* out = (float*)d_out;                               // T*B*E
  float* avg = out + (size_t)T_DIM * B_DIM * E_DIM;         // B*T*T

  const size_t headElems = (size_t)B_DIM * H_DIM * T_DIM * HD_DIM;  // 6.29M
  const size_t xElems = (size_t)M_DIM * E_DIM;                      // 6.29M
  _Float16* Qh  = (_Float16*)d_ws;
  _Float16* Kh  = Qh + headElems;
  _Float16* Vt  = Kh + headElems;
  _Float16* stg = Vt + headElems;        // f16 attn staging [T,B,E]
  _Float16* Xq  = stg + xElems;
  _Float16* Xk  = Xq + xElems;
  _Float16* Xv  = Xk + xElems;
  _Float16* Wh  = Xv + xElems;           // 3E x E
  _Float16* Woh = Wh + (size_t)3 * E_DIM * E_DIM;

  // ---- pre-pass: convert operands to f16 once (bandwidth-bound) ----
  {
    const int nq4 = (int)(xElems / 4);
    const int nw4 = (int)((size_t)3 * E_DIM * E_DIM / 4);
    const int no4 = (int)((size_t)E_DIM * E_DIM / 4);
    cvt_f16_kernel<<<2048, 256, 0, stream>>>(query, Xq, nq4);
    cvt_f16_kernel<<<2048, 256, 0, stream>>>(key, Xk, nq4);
    cvt_f16_kernel<<<2048, 256, 0, stream>>>(value, Xv, nq4);
    cvt_f16_kernel<<<1024, 256, 0, stream>>>(w_in, Wh, nw4);
    cvt_f16_kernel<<<512, 256, 0, stream>>>(w_out, Woh, no4);
  }

  dim3 blk(128);
  qkv_proj_kernel<<<dim3(M_DIM / 32, 9), blk, 0, stream>>>(
      Xq, Xk, Xv, Wh, b_in, Qh, Kh, Vt);

  const size_t smem = (16 * 2048 + 4 * 2 * 16 * 17) * sizeof(float); // ~137 KB
  attention_kernel<<<dim3(T_DIM / 16, B_DIM), blk, smem, stream>>>(
      Qh, Kh, Vt, stg, avg);

  out_proj_kernel<<<dim3(M_DIM / 32, 3), blk, 0, stream>>>(
      stg, Woh, b_out, out);
}
